// _AttnBlock_36043365548384
// MI455X (gfx1250) — matrix operand order repaired and checked by reading, not compiled, not run on hardware
//
#include <hip/hip_runtime.h>
#include <hip/hip_bf16.h>

// ---------------------------------------------------------------------------
// XCA (cross-covariance attention) block for MI455X / gfx1250.
// All 4 GEMMs on v_wmma_f32_16x16x32_f16 (f16 in, f32 accum).
// Layouts: q,k channel-major [c][n]; v,O token-major [n][c]; attn [c][d];
// weights pre-converted to f16 once. Every WMMA fragment in every K-loop is
// a contiguous 16B/32B vector load. The S GEMM (K=4096) is split into 8
// K-chunks (256 blocks) with f32 partial tiles; a reduce kernel applies the
// bilinear l2-norm/temperature scaling and does the row softmax.
// ---------------------------------------------------------------------------

typedef _Float16 h16;
typedef __attribute__((ext_vector_type(16))) _Float16 v16h;
typedef __attribute__((ext_vector_type(8)))  _Float16 v8h;
typedef __attribute__((ext_vector_type(8)))  float    v8f;

#define GROUPS  16
#define BATCH   8
#define CCH     256
#define NPIX    4096            // 64*64 tokens
#define KCHUNK  512             // S-GEMM K split
#define NCHUNKS (NPIX / KCHUNK) // 8
#define EPS_GN  1e-6f
#define EPS_L2  1e-12f
#define LDSA_LD 264             // padded LDS row (halves), conflict-free A reads

static __device__ __forceinline__ v8f wmma16x16x32(v16h a, v16h b, v8f c) {
  return __builtin_amdgcn_wmma_f32_16x16x32_f16(false, a, false, b, (short)0, c,
                                                false, false);
}

// A fragment: elements 0..7 at p[0..7], elements 8..15 at p[8..15]
// (p already includes row*ld + klo, klo = (lane>>4)*16).
static __device__ __forceinline__ v16h load_a_frag(const h16* p) {
  v16h a;
  *(v8h*)&a       = *(const v8h*)p;
  *((v8h*)&a + 1) = *(const v8h*)(p + 8);
  return a;
}

// -------------------------------- utility ---------------------------------

__global__ __launch_bounds__(256) void k_zero(float* p, int n) {
  int i = blockIdx.x * 256 + threadIdx.x;
  if (i < n) p[i] = 0.0f;
}

// One-time f32 -> f16 conversion of qkv_w (768x256) and proj_w (256x256).
__global__ __launch_bounds__(256) void k_wconv(const float* __restrict__ qkv_w,
                                               const float* __restrict__ proj_w,
                                               h16* __restrict__ w_h) {
  int i = blockIdx.x * 256 + threadIdx.x;
  if (i < 768 * 256)
    w_h[i] = (h16)qkv_w[i];
  else if (i < 768 * 256 + 256 * 256)
    w_h[i] = (h16)proj_w[i - 768 * 256];
}

// ------------------------- GroupNorm statistics ----------------------------
__global__ __launch_bounds__(256) void k_gnstats(const float* __restrict__ x,
                                                 float* __restrict__ gn_mean,
                                                 float* __restrict__ gn_rstd) {
  __shared__ float s_sum[256];
  __shared__ float s_sq[256];
  int b = blockIdx.x >> 4, g = blockIdx.x & 15;
  const float4* base = (const float4*)(x + ((size_t)b * CCH + g * 16) * NPIX);
  float s = 0.f, q = 0.f;
  for (int e = threadIdx.x; e < 16 * NPIX / 4; e += 256) {
    float4 v = base[e];
    s += v.x + v.y + v.z + v.w;
    q += v.x * v.x + v.y * v.y + v.z * v.z + v.w * v.w;
  }
  s_sum[threadIdx.x] = s; s_sq[threadIdx.x] = q;
  __syncthreads();
  for (int st = 128; st > 0; st >>= 1) {
    if (threadIdx.x < (unsigned)st) {
      s_sum[threadIdx.x] += s_sum[threadIdx.x + st];
      s_sq[threadIdx.x]  += s_sq[threadIdx.x + st];
    }
    __syncthreads();
  }
  if (threadIdx.x == 0) {
    float m   = s_sum[0] * (1.0f / 65536.0f);
    float var = s_sq[0] * (1.0f / 65536.0f) - m * m;
    gn_mean[blockIdx.x] = m;
    gn_rstd[blockIdx.x] = rsqrtf(var + EPS_GN);
  }
}

// -------------------- fused GN -> QKV GEMM (WMMA) --------------------------
// q,k -> channel-major [b][2][C][N]; v -> token-major [b][N][C].
__global__ __launch_bounds__(256) void k_qkv(const float* __restrict__ x,
                                             const float* __restrict__ gamma,
                                             const float* __restrict__ beta,
                                             const h16* __restrict__ qkv_wh,
                                             const float* __restrict__ qkv_b,
                                             const float* __restrict__ gn_mean,
                                             const float* __restrict__ gn_rstd,
                                             h16* __restrict__ qk_h,
                                             h16* __restrict__ v_t,
                                             float* __restrict__ l2sum) {
  __shared__ h16 lds_a[64 * LDSA_LD];   // normalized [token][channel], padded
  int b  = blockIdx.z;
  int t0 = blockIdx.x * 64;             // token tile base
  int j0 = blockIdx.y * 64;             // output-column tile base
  for (int e = threadIdx.x; e < 64 * 256; e += 256) {
    int c = e >> 6;
    int n = e & 63;
    float xv = x[((size_t)b * CCH + c) * NPIX + t0 + n];
    int g = c >> 4;
    float hv = (xv - gn_mean[b * GROUPS + g]) * gn_rstd[b * GROUPS + g]
               * gamma[c] + beta[c];
    lds_a[n * LDSA_LD + c] = (h16)hv;
  }
  __syncthreads();

  int wv   = threadIdx.x >> 5;
  int lane = threadIdx.x & 31;
  int m0 = (wv & 3) * 16;
  int n0 = (wv >> 2) * 32;
  int arow = m0 + (lane & 15);
  int aklo = (lane >> 4) << 4;
  int bcol = lane & 15;
  int bklo = (lane >> 4) << 4;
  const h16* abase = lds_a + arow * LDSA_LD + aklo;
  const h16* w0 = qkv_wh + (size_t)(j0 + n0 + bcol) * 256 + bklo;
  const h16* w1 = qkv_wh + (size_t)(j0 + n0 + 16 + bcol) * 256 + bklo;

  v8f acc[2] = {v8f{}, v8f{}};
#pragma unroll
  for (int k0 = 0; k0 < 256; k0 += 32) {
    v16h af  = load_a_frag(abase + k0);
    v16h bf0 = *(const v16h*)(w0 + k0);
    v16h bf1 = *(const v16h*)(w1 + k0);
    acc[0] = wmma16x16x32(af, bf0, acc[0]);
    acc[1] = wmma16x16x32(af, bf1, acc[1]);
  }

  int mrow = (lane >> 4) << 3;
#pragma unroll
  for (int f = 0; f < 2; ++f) {
    int j     = j0 + n0 + f * 16 + bcol;   // 0..767
    int which = j >> 8;                    // 0=q 1=k 2=v
    int c     = j & 255;
    float bias = qkv_b[j];
    float ss = 0.f;
    v8h pk;
#pragma unroll
    for (int r = 0; r < 8; ++r) {
      float vv = acc[f][r] + bias;
      ss += vv * vv;
      pk[r] = (h16)vv;
    }
    if (which < 2) {
      // channel-major, 8 contiguous tokens -> single 16B store
      h16* dst = qk_h + (((size_t)b * 2 + which) * CCH + c) * NPIX;
      *(v8h*)(dst + t0 + m0 + mrow) = pk;
      atomicAdd(&l2sum[(b * 2 + which) * CCH + c], ss);
    } else {
      // v: token-major
#pragma unroll
      for (int r = 0; r < 8; ++r)
        v_t[((size_t)b * NPIX + t0 + m0 + mrow + r) * CCH + c] = pk[r];
    }
  }
}

// ------------------------- finalize inverse L2 norms -----------------------
__global__ __launch_bounds__(256) void k_l2fin(float* l2) {
  int i = blockIdx.x * 256 + threadIdx.x;
  if (i < BATCH * 2 * CCH) {
    float n = sqrtf(l2[i]);
    l2[i] = 1.0f / fmaxf(n, EPS_L2);
  }
}

// ----------------- S partials: raw q k^T over one K-chunk ------------------
// Grid (4 row-tiles, 8 K-chunks, 8 batches) = 256 blocks for parallelism.
// Pure loads + WMMA in the loop; f32 partial tile to scratch.
__global__ __launch_bounds__(256) void k_attn_part(const h16* __restrict__ qk_h,
                                                   float* __restrict__ spart) {
  int b  = blockIdx.z;
  int c0 = blockIdx.x * 64;
  int kbase = blockIdx.y * KCHUNK;
  const h16* qb = qk_h + ((size_t)b * 2 + 0) * CCH * NPIX;
  const h16* kb = qk_h + ((size_t)b * 2 + 1) * CCH * NPIX;

  int wv   = threadIdx.x >> 5;
  int lane = threadIdx.x & 31;
  int m0 = (wv >> 1) * 16;
  int n0 = (wv & 1) * 128;
  int arow = c0 + m0 + (lane & 15);
  int aklo = (lane >> 4) << 4;
  int lcol = lane & 15;
  int bklo = (lane >> 4) << 4;
  const h16* qrow = qb + (size_t)arow * NPIX + kbase + aklo;

  v8f acc[8] = {};
  for (int k0 = 0; k0 < KCHUNK; k0 += 32) {
    v16h af = load_a_frag(qrow + k0);
#pragma unroll
    for (int f = 0; f < 8; ++f) {
      int col = n0 + f * 16 + lcol;
      v16h bf = *(const v16h*)(kb + (size_t)col * NPIX + kbase + k0 + bklo);
      acc[f] = wmma16x16x32(af, bf, acc[f]);
    }
  }

  // partial tile: [b][ctile][chunk][64][256] f32
  float* sp = spart +
      ((((size_t)b * 4 + blockIdx.x) * NCHUNKS + blockIdx.y) << 14);
  int mrow = (lane >> 4) << 3;
#pragma unroll
  for (int f = 0; f < 8; ++f) {
    int col = n0 + f * 16 + lcol;
#pragma unroll
    for (int r = 0; r < 8; ++r)
      sp[(m0 + mrow + r) * 256 + col] = acc[f][r];
  }
}

// --------- reduce partials, apply invq*invk*temperature, softmax -----------
__global__ __launch_bounds__(256) void k_softmax(const float* __restrict__ spart,
                                                 const float* __restrict__ l2inv,
                                                 const float* __restrict__ temperature,
                                                 h16* __restrict__ attn_h) {
  __shared__ float lds_s[64 * 256];   // 64 KB scaled-logits tile
  int b  = blockIdx.z;
  int c0 = blockIdx.x * 64;
  const float* qinv = l2inv + (b * 2 + 0) * CCH;
  const float* kinv = l2inv + (b * 2 + 1) * CCH;
  float temp = temperature[0];
  const float* sp = spart + ((((size_t)b * 4 + blockIdx.x) * NCHUNKS) << 14);

  for (int e = threadIdx.x; e < 64 * 256; e += 256) {
    float s = 0.f;
#pragma unroll
    for (int ch = 0; ch < NCHUNKS; ++ch) s += sp[(ch << 14) + e];
    int row = e >> 8, col = e & 255;
    lds_s[e] = s * (qinv[c0 + row] * temp) * kinv[col];
  }
  __syncthreads();

  // Row softmax over 256 cols: 4 threads per row + shfl butterflies.
  int r  = threadIdx.x >> 2;          // row 0..63
  int qd = threadIdx.x & 3;           // quarter
  float* row = &lds_s[r * 256 + qd * 64];
  float mx = row[0];
  for (int i = 1; i < 64; ++i) mx = fmaxf(mx, row[i]);
  mx = fmaxf(mx, __shfl_xor(mx, 1));
  mx = fmaxf(mx, __shfl_xor(mx, 2));
  float s = 0.f;
  for (int i = 0; i < 64; ++i) { float e = expf(row[i] - mx); row[i] = e; s += e; }
  s += __shfl_xor(s, 1);
  s += __shfl_xor(s, 2);
  float inv = 1.0f / s;
  h16* dst = attn_h + ((size_t)b * CCH + c0 + r) * 256 + qd * 64;
#pragma unroll
  for (int j8 = 0; j8 < 8; ++j8) {
    v8h pk;
#pragma unroll
    for (int jj = 0; jj < 8; ++jj) pk[jj] = (h16)(row[j8 * 8 + jj] * inv);
    *(v8h*)(dst + j8 * 8) = pk;
  }
}

// ------------------ O^T = v^T attn^T  (token-major output) -----------------
// O_t[n][c] = sum_d v_t[n][d] * attn[c][d].  M=tokens, N=c, K=d.
__global__ __launch_bounds__(256) void k_av(const h16* __restrict__ attn_h,
                                            const h16* __restrict__ v_t,
                                            h16* __restrict__ o_t) {
  int b  = blockIdx.z;
  int t0 = blockIdx.x * 64;             // token tile (block covers all 256 c)
  const h16* ab = attn_h + (size_t)b * CCH * 256;
  const h16* vb = v_t + (size_t)b * NPIX * CCH;

  int wv   = threadIdx.x >> 5;
  int lane = threadIdx.x & 31;
  int m0 = (wv >> 1) * 16;              // token sub-tile
  int n0 = (wv & 1) * 128;              // channel sub-tile
  int arow = t0 + m0 + (lane & 15);
  int aklo = (lane >> 4) << 4;
  int lcol = lane & 15;
  int bklo = (lane >> 4) << 4;
  const h16* vrow = vb + (size_t)arow * CCH + aklo;

  v8f acc[8] = {};
#pragma unroll
  for (int k0 = 0; k0 < 256; k0 += 32) {
    v16h af = load_a_frag(vrow + k0);
#pragma unroll
    for (int f = 0; f < 8; ++f) {
      int col = n0 + f * 16 + lcol;     // output channel
      v16h bf = *(const v16h*)(ab + (size_t)col * 256 + k0 + bklo);
      acc[f] = wmma16x16x32(af, bf, acc[f]);
    }
  }

  int mrow = (lane >> 4) << 3;
#pragma unroll
  for (int f = 0; f < 8; ++f) {
    int col = n0 + f * 16 + lcol;
#pragma unroll
    for (int r = 0; r < 8; ++r)
      o_t[((size_t)b * NPIX + t0 + m0 + mrow + r) * CCH + col] = (h16)acc[f][r];
  }
}

// ------------------- out = proj(O) + bias + residual -----------------------
// final[b][co][n] = x[b][co][n] + proj_b[co] + sum_c proj_w[co][c] * O_t[n][c]
__global__ __launch_bounds__(256) void k_proj(const h16* __restrict__ o_t,
                                              const h16* __restrict__ proj_wh,
                                              const float* __restrict__ proj_b,
                                              const float* __restrict__ x,
                                              float* __restrict__ out) {
  int b  = blockIdx.z;
  int c0 = blockIdx.x * 64;             // output-channel tile
  int t0 = blockIdx.y * 256;            // token tile
  const h16* ob = o_t + (size_t)b * NPIX * CCH;

  int wv   = threadIdx.x >> 5;
  int lane = threadIdx.x & 31;
  int m0 = (wv >> 1) * 16;
  int n0 = (wv & 1) * 128;
  int arow = c0 + m0 + (lane & 15);
  int aklo = (lane >> 4) << 4;
  int lcol = lane & 15;
  int bklo = (lane >> 4) << 4;
  const h16* wp = proj_wh + (size_t)arow * 256 + aklo;

  v8f acc[8] = {};
#pragma unroll
  for (int k0 = 0; k0 < 256; k0 += 32) {
    v16h af = load_a_frag(wp + k0);
#pragma unroll
    for (int f = 0; f < 8; ++f) {
      int tcol = t0 + n0 + f * 16 + lcol;
      v16h bf = *(const v16h*)(ob + (size_t)tcol * 256 + k0 + bklo);
      acc[f] = wmma16x16x32(af, bf, acc[f]);
    }
  }

  int mrow = (lane >> 4) << 3;
#pragma unroll
  for (int f = 0; f < 8; ++f) {
    int tcol = t0 + n0 + f * 16 + lcol;
#pragma unroll
    for (int r = 0; r < 8; ++r) {
      int co = c0 + m0 + mrow + r;
      size_t idx = ((size_t)b * CCH + co) * NPIX + tcol;
      out[idx] = x[idx] + proj_b[co] + acc[f][r];
    }
  }
}

// ---------------------------------------------------------------------------

extern "C" void kernel_launch(void* const* d_in, const int* in_sizes, int n_in,
                              void* d_out, int out_size, void* d_ws, size_t ws_size,
                              hipStream_t stream) {
  const float* x       = (const float*)d_in[0];
  const float* gamma   = (const float*)d_in[1];
  const float* beta    = (const float*)d_in[2];
  const float* qkv_w   = (const float*)d_in[3];
  const float* qkv_b   = (const float*)d_in[4];
  const float* temp    = (const float*)d_in[5];
  const float* proj_w  = (const float*)d_in[6];
  const float* proj_b  = (const float*)d_in[7];
  float* out = (float*)d_out;

  // Workspace layout
  float* ws_f    = (float*)d_ws;
  float* gn_mean = ws_f;               // 128
  float* gn_rstd = ws_f + 128;         // 128
  float* l2sum   = ws_f + 256;         // 8*2*256 = 4096
  h16* w_h     = (h16*)((char*)d_ws + 32768);           // qkv_w | proj_w f16
  h16* qkv_wh  = w_h;                                   // 768*256
  h16* proj_wh = w_h + 768 * 256;                       // 256*256
  h16* qk_h   = w_h + (768 + 256) * 256;                // q,k: [b][2][C][N]
  h16* v_t    = qk_h + (size_t)BATCH * 2 * CCH * NPIX;  // v:   [b][N][C]
  h16* attn_h = v_t + (size_t)BATCH * NPIX * CCH;       // [b][C][C]
  h16* o_t    = attn_h + (size_t)BATCH * CCH * 256;     // [b][N][C]
  float* spart = (float*)(o_t + (size_t)BATCH * NPIX * CCH); // [b][4][8][64][256]

  k_zero<<<16, 256, 0, stream>>>(l2sum, BATCH * 2 * CCH);
  k_wconv<<<(768 * 256 + 256 * 256) / 256, 256, 0, stream>>>(qkv_w, proj_w, w_h);
  k_gnstats<<<BATCH * GROUPS, 256, 0, stream>>>(x, gn_mean, gn_rstd);
  k_qkv<<<dim3(NPIX / 64, 768 / 64, BATCH), 256, 0, stream>>>(
      x, gamma, beta, qkv_wh, qkv_b, gn_mean, gn_rstd, qk_h, v_t, l2sum);
  k_l2fin<<<16, 256, 0, stream>>>(l2sum);
  k_attn_part<<<dim3(CCH / 64, NCHUNKS, BATCH), 256, 0, stream>>>(qk_h, spart);
  k_softmax<<<dim3(CCH / 64, 1, BATCH), 256, 0, stream>>>(spart, l2sum, temp, attn_h);
  k_av<<<dim3(NPIX / 64, 1, BATCH), 256, 0, stream>>>(attn_h, v_t, o_t);
  k_proj<<<dim3(CCH / 64, NPIX / 256, BATCH), 256, 0, stream>>>(
      o_t, proj_wh, proj_b, x, out);
}
